// MultiHeadAttention_16295105921164
// MI455X (gfx1250) — matrix operand order repaired and checked by reading, not compiled, not run on hardware
//
#include <hip/hip_runtime.h>
#include <hip/hip_bf16.h>
#include <stdint.h>

typedef __attribute__((ext_vector_type(16))) __bf16 v16bf;
typedef __attribute__((ext_vector_type(8)))  __bf16 bf16x8;
typedef __attribute__((ext_vector_type(8)))  float  v8f;

#define DIMN  1024
#define LSEQ  2048
#define BATCH 2
#define NH    16
#define HEADD 64
#define MROWS (BATCH*LSEQ)   /* 4096 */
#define LOG2E 1.4426950408889634f
#define BK    32
#define PITCH 40             /* 32 data + 8 pad bf16 per LDS tile row (80B pitch) */

#if defined(__has_builtin)
#if __has_builtin(__builtin_amdgcn_tensor_load_to_lds) && __has_builtin(__builtin_amdgcn_s_wait_tensorcnt)
#define HAVE_TDM 1
#endif
#endif
#ifndef HAVE_TDM
#define HAVE_TDM 0
#endif

__device__ __forceinline__ unsigned short f2bf(float f){
  union { float f; uint32_t u; } x; x.f = f;
  uint32_t u = x.u;
  uint32_t r = u + 0x7FFFu + ((u >> 16) & 1u);   // round-to-nearest-even
  return (unsigned short)(r >> 16);
}

__device__ __forceinline__ v16bf combine8(bf16x8 lo, bf16x8 hi){
  v16bf r;
#pragma unroll
  for (int t = 0; t < 8; ++t){ r[t] = lo[t]; r[t+8] = hi[t]; }
  return r;
}

__device__ __forceinline__ v8f zero8(){
  v8f z;
#pragma unroll
  for (int t = 0; t < 8; ++t) z[t] = 0.0f;
  return z;
}

#if HAVE_TDM
typedef unsigned int u32x4 __attribute__((ext_vector_type(4)));
typedef int          i32x4 __attribute__((ext_vector_type(4)));
typedef int          i32x8 __attribute__((ext_vector_type(8)));

__device__ __forceinline__ unsigned lds_off(const void* p){
  // LDS aperture: flat addr low 32 bits are the LDS byte offset (ISA 10.2)
  return (unsigned)(uintptr_t)p;
}

// 2D bf16 tile (tile_d0=BK elems x tile_d1=128 rows) Global -> LDS, with LDS
// padding of 4 DWORDs every 16 DWORDs (one 64B row) => 80B LDS row pitch.
__device__ __forceinline__ void tdm_load_tile_2d(const unsigned short* g, unsigned ldsoff,
                                                 int tensor_d1){
  unsigned long long ga = (unsigned long long)(uintptr_t)g;
  u32x4 g0;
  g0[0] = 1u;                                            // count=1 (valid), user mode
  g0[1] = ldsoff;                                        // lds_addr [63:32]
  g0[2] = (unsigned)(ga & 0xFFFFFFFFu);                  // global_addr lo
  g0[3] = ((unsigned)(ga >> 32) & 0x01FFFFFFu) | (2u << 30); // global_addr hi | type=2
  i32x8 g1;
  g1[0] = (1 << 16)                                      // data_size=1 -> 2 bytes
        | (1 << 20)                                      // pad_enable
        | (3 << 22)                                      // pad_interval: 16 DWORDs
        | (3 << 25);                                     // pad_amount: 4 DWORDs (16B)
  g1[1] = (int)(((unsigned)DIMN & 0xFFFFu) << 16);       // tensor_dim0 lo16 @ [79:48]
  g1[2] = (int)((((unsigned)DIMN >> 16) & 0xFFFFu)       // tensor_dim0 hi16
        | (((unsigned)tensor_d1 & 0xFFFFu) << 16));      // tensor_dim1 lo16
  g1[3] = (int)((((unsigned)tensor_d1 >> 16) & 0xFFFFu)  // tensor_dim1 hi16
        | ((unsigned)BK << 16));                         // tile_dim0 @ [127:112]
  g1[4] = 128;                                           // tile_dim1 @ [143:128]
  g1[5] = DIMN;                                          // tensor_dim0_stride lo32
  g1[6] = 0;                                             // stride hi16 | dim1_stride lo16
  g1[7] = 0;
  i32x4 z4 = {0, 0, 0, 0};
#if defined(__clang_major__) && __clang_major__ >= 23
  i32x8 z8 = {0, 0, 0, 0, 0, 0, 0, 0};
  __builtin_amdgcn_tensor_load_to_lds(g0, g1, z4, z4, z8, 0);
#else
  __builtin_amdgcn_tensor_load_to_lds(g0, g1, z4, z4, 0);
#endif
}
#endif

// ---------------------------------------------------------------- cvt f32->bf16
__global__ __launch_bounds__(256) void cvt_bf16(const float* __restrict__ src,
                                                unsigned short* __restrict__ dst,
                                                int n, float scale){
  for (int i = blockIdx.x*blockDim.x + threadIdx.x; i < n; i += gridDim.x*blockDim.x)
    dst[i] = f2bf(src[i] * scale);
}

// ---------------------------------------------------------------- NT GEMM via WMMA
// C(4096x1024) = A(4096x1024 bf16 rm) * B(1024x1024 bf16 rm)^T
// 128x128 block tile, K=32 steps, LDS double-buffered, filled by TDM.
// mode 0: f32 row-major; 1: bf16 row-major; 2: bf16 per-head-transposed (V)
__global__ __launch_bounds__(256) void gemm_nt_wmma(const unsigned short* __restrict__ A,
                                                    const unsigned short* __restrict__ Bm,
                                                    float* __restrict__ Cf,
                                                    unsigned short* __restrict__ Cb,
                                                    int mode){
  __shared__ unsigned short Asm[2][128*PITCH];
  __shared__ unsigned short Bsm[2][128*PITCH];

  const int tid  = threadIdx.x;
  const int lane = tid & 31;
  const int lr   = lane & 15;
  const int sub  = lane >> 4;
  const int wave = __builtin_amdgcn_readfirstlane((int)tid) >> 5;  // uniform wave id
  const int wm   = wave >> 1;                  // 0..3 : 32-row stripe
  const int wn   = wave & 1;                   // 0..1 : 64-col stripe
  const int rowBase = blockIdx.y * 128;
  const int colBase = blockIdx.x * 128;

  v8f acc[2][4];
#pragma unroll
  for (int mi = 0; mi < 2; ++mi)
#pragma unroll
    for (int nt = 0; nt < 4; ++nt) acc[mi][nt] = zero8();

  auto load_tiles = [&](int buf, int kk){
#if HAVE_TDM
    if (wave == 0)
      tdm_load_tile_2d(A  + (size_t)rowBase*DIMN + kk, lds_off(&Asm[buf][0]), MROWS);
    else if (wave == 1)
      tdm_load_tile_2d(Bm + (size_t)colBase*DIMN + kk, lds_off(&Bsm[buf][0]), DIMN);
#else
#pragma unroll
    for (int s2 = tid; s2 < 512; s2 += 256){
      int r = s2 >> 2;
      int c = (s2 & 3) * 8;
      *(uint4*)&Asm[buf][r*PITCH + c] = *(const uint4*)(A  + (size_t)(rowBase + r)*DIMN + kk + c);
      *(uint4*)&Bsm[buf][r*PITCH + c] = *(const uint4*)(Bm + (size_t)(colBase + r)*DIMN + kk + c);
    }
#endif
  };

  load_tiles(0, 0);
#if HAVE_TDM
  __builtin_amdgcn_s_wait_tensorcnt((short)0);
#endif
  __syncthreads();

  for (int it = 0; it < DIMN/BK; ++it){
    const int cur = it & 1;
    if (it + 1 < DIMN/BK) load_tiles(cur ^ 1, (it + 1)*BK);

    v16bf af[2];
#pragma unroll
    for (int mi = 0; mi < 2; ++mi){
      const unsigned short* p = &Asm[cur][(wm*32 + mi*16 + lr)*PITCH + sub*8];
      af[mi] = combine8(*(const bf16x8*)p, *(const bf16x8*)(p + 16));
    }
    v16bf bfr[4];
#pragma unroll
    for (int nt = 0; nt < 4; ++nt){
      const unsigned short* p = &Bsm[cur][(wn*64 + nt*16 + lr)*PITCH + sub*8];
      bfr[nt] = combine8(*(const bf16x8*)p, *(const bf16x8*)(p + 16));
    }
#pragma unroll
    for (int mi = 0; mi < 2; ++mi)
#pragma unroll
      for (int nt = 0; nt < 4; ++nt)
        acc[mi][nt] = __builtin_amdgcn_wmma_f32_16x16x32_bf16(false, af[mi], false, bfr[nt],
                                                              (short)0, acc[mi][nt], false, false);

#if HAVE_TDM
    __builtin_amdgcn_s_wait_tensorcnt((short)0);
#endif
    __syncthreads();
  }

#pragma unroll
  for (int mi = 0; mi < 2; ++mi)
#pragma unroll
    for (int nt = 0; nt < 4; ++nt){
      v8f av = acc[mi][nt];
#pragma unroll
      for (int r = 0; r < 8; ++r){
        int row = rowBase + wm*32 + mi*16 + r + sub*8;
        int col = colBase + wn*64 + nt*16 + lr;
        float val = av[r];
        if (mode == 0){
          Cf[(size_t)row*DIMN + col] = val;
        } else if (mode == 1){
          Cb[(size_t)row*DIMN + col] = f2bf(val);
        } else {
          int b  = row >> 11;
          int rl = row & 2047;
          int n  = rl >> 7;
          int i  = ((rl & 127) << 4) + (col >> 6);
          int d  = col & 63;
          Cb[((size_t)((b*NH + n)*HEADD + d))*LSEQ + i] = f2bf(val);
        }
      }
    }
}

// ---------------------------------------------------------------- flash attention
__global__ __launch_bounds__(32) void flash_attn(const unsigned short* __restrict__ Qp,
                                                 const unsigned short* __restrict__ Kp,
                                                 const unsigned short* __restrict__ Vt,
                                                 unsigned short* __restrict__ Ctx){
  __shared__ unsigned short Pbuf[32*32];
  const int lane = threadIdx.x;
  const int lr   = lane & 15;
  const int sub  = lane >> 4;
  const int h = blockIdx.y;           // b*16+n
  const int b = h >> 4;
  const int n = h & 15;
  const int itile = blockIdx.x * 32;

  const unsigned short* Qb = Qp + (size_t)h * LSEQ * HEADD;
  const unsigned short* Kb = Kp + (size_t)h * LSEQ * HEADD;
  const unsigned short* Vb = Vt + (size_t)h * HEADD * LSEQ;

  v16bf qa[2][2];
#pragma unroll
  for (int mi = 0; mi < 2; ++mi)
#pragma unroll
    for (int ks = 0; ks < 2; ++ks){
      const unsigned short* p = Qb + (size_t)(itile + mi*16 + lr)*HEADD + ks*32 + sub*8;
      qa[mi][ks] = combine8(*(const bf16x8*)p, *(const bf16x8*)(p + 16));
    }

  float mstate[2][8], lstate[2][8];
  v8f o[2][4];
#pragma unroll
  for (int mi = 0; mi < 2; ++mi){
#pragma unroll
    for (int r = 0; r < 8; ++r){ mstate[mi][r] = -3.0e38f; lstate[mi][r] = 0.0f; }
#pragma unroll
    for (int dt = 0; dt < 4; ++dt) o[mi][dt] = zero8();
  }

  for (int jt = 0; jt < LSEQ/32; ++jt){
    const int jb = jt*32;

    v16bf kf[2][2];
#pragma unroll
    for (int nj = 0; nj < 2; ++nj)
#pragma unroll
      for (int ks = 0; ks < 2; ++ks)
        kf[nj][ks] = combine8(*(const bf16x8*)(Kb + (size_t)(jb + nj*16 + lr)*HEADD + ks*32 + sub*8), *(const bf16x8*)(Kb + (size_t)(jb + nj*16 + lr)*HEADD + ks*32 + sub*8 + 16));

    float s[2][2][8];
#pragma unroll
    for (int mi = 0; mi < 2; ++mi)
#pragma unroll
      for (int nj = 0; nj < 2; ++nj){
        v8f sv = zero8();
        sv = __builtin_amdgcn_wmma_f32_16x16x32_bf16(false, qa[mi][0], false, kf[nj][0], (short)0, sv, false, false);
        sv = __builtin_amdgcn_wmma_f32_16x16x32_bf16(false, qa[mi][1], false, kf[nj][1], (short)0, sv, false, false);
#pragma unroll
        for (int r = 0; r < 8; ++r) s[mi][nj][r] = sv[r];
      }

    __syncthreads();   // previous-iteration Pbuf reads complete

#pragma unroll
    for (int mi = 0; mi < 2; ++mi){
      float alpha[8];
#pragma unroll
      for (int r = 0; r < 8; ++r){
        float rm = fmaxf(s[mi][0][r], s[mi][1][r]);
        rm = fmaxf(rm, __shfl_xor(rm, 1, 32));
        rm = fmaxf(rm, __shfl_xor(rm, 2, 32));
        rm = fmaxf(rm, __shfl_xor(rm, 4, 32));
        rm = fmaxf(rm, __shfl_xor(rm, 8, 32));
        float mo = mstate[mi][r];
        float mn = fmaxf(mo, rm);
        alpha[r] = exp2f((mo - mn) * LOG2E);
        mstate[mi][r] = mn;
        float p0 = exp2f((s[mi][0][r] - mn) * LOG2E);
        float p1 = exp2f((s[mi][1][r] - mn) * LOG2E);
        s[mi][0][r] = p0; s[mi][1][r] = p1;
        float rs = p0 + p1;
        rs += __shfl_xor(rs, 1, 32);
        rs += __shfl_xor(rs, 2, 32);
        rs += __shfl_xor(rs, 4, 32);
        rs += __shfl_xor(rs, 8, 32);
        lstate[mi][r] = lstate[mi][r] * alpha[r] + rs;
      }
#pragma unroll
      for (int dt = 0; dt < 4; ++dt)
#pragma unroll
        for (int r = 0; r < 8; ++r) o[mi][dt][r] *= alpha[r];
#pragma unroll
      for (int nj = 0; nj < 2; ++nj)
#pragma unroll
        for (int r = 0; r < 8; ++r)
          Pbuf[(mi*16 + r + sub*8)*32 + nj*16 + lr] = f2bf(s[mi][nj][r]);
    }

    __syncthreads();   // P visible to all lanes

    v16bf pa[2];
#pragma unroll
    for (int mi = 0; mi < 2; ++mi){
      const unsigned short* pp = &Pbuf[(mi*16 + lr)*32 + sub*8];
      pa[mi] = combine8(*(const bf16x8*)pp, *(const bf16x8*)(pp + 16));
    }
    v16bf vbf[4];
#pragma unroll
    for (int dt = 0; dt < 4; ++dt)
      vbf[dt] = combine8(*(const bf16x8*)(Vb + (size_t)(dt*16 + lr)*LSEQ + jb + sub*8), *(const bf16x8*)(Vb + (size_t)(dt*16 + lr)*LSEQ + jb + sub*8 + 16));
#pragma unroll
    for (int mi = 0; mi < 2; ++mi)
#pragma unroll
      for (int dt = 0; dt < 4; ++dt)
        o[mi][dt] = __builtin_amdgcn_wmma_f32_16x16x32_bf16(false, pa[mi], false, vbf[dt], (short)0, o[mi][dt], false, false);
  }

#pragma unroll
  for (int mi = 0; mi < 2; ++mi)
#pragma unroll
    for (int dt = 0; dt < 4; ++dt)
#pragma unroll
      for (int r = 0; r < 8; ++r){
        float val = o[mi][dt][r] / lstate[mi][r];
        int row = b*LSEQ + itile + mi*16 + r + sub*8;
        int col = n*HEADD + dt*16 + lr;
        Ctx[(size_t)row*DIMN + col] = f2bf(val);
      }
}

// ---------------------------------------------------------------- host launcher
extern "C" void kernel_launch(void* const* d_in, const int* in_sizes, int n_in,
                              void* d_out, int out_size, void* d_ws, size_t ws_size,
                              hipStream_t stream){
  const float* q  = (const float*)d_in[0];
  const float* k  = (const float*)d_in[1];
  const float* v  = (const float*)d_in[2];
  const float* Wq = (const float*)d_in[3];
  const float* Wk = (const float*)d_in[4];
  const float* Wv = (const float*)d_in[5];
  const float* Wo = (const float*)d_in[6];
  float* out = (float*)d_out;

  const size_t NX = (size_t)MROWS * DIMN;   // 4M elems
  const size_t NW = (size_t)DIMN * DIMN;    // 1M elems
  unsigned short* Xq  = (unsigned short*)d_ws;
  unsigned short* Xk  = Xq  + NX;
  unsigned short* Xv  = Xk  + NX;
  unsigned short* Wqb = Xv  + NX;
  unsigned short* Wkb = Wqb + NW;
  unsigned short* Wvb = Wkb + NW;
  unsigned short* Wob = Wvb + NW;
  unsigned short* Qp  = Wob + NW;
  unsigned short* Kp  = Qp  + NX;
  unsigned short* Vt  = Kp  + NX;
  unsigned short* Ctx = Vt  + NX;

  dim3 cB(256), cG(1024);
  cvt_bf16<<<cG, cB, 0, stream>>>(q,  Xq,  (int)NX, 1.0f);
  cvt_bf16<<<cG, cB, 0, stream>>>(k,  Xk,  (int)NX, 1.0f);
  cvt_bf16<<<cG, cB, 0, stream>>>(v,  Xv,  (int)NX, 1.0f);
  cvt_bf16<<<cG, cB, 0, stream>>>(Wq, Wqb, (int)NW, 0.125f);  // fold 1/sqrt(64)
  cvt_bf16<<<cG, cB, 0, stream>>>(Wk, Wkb, (int)NW, 1.0f);
  cvt_bf16<<<cG, cB, 0, stream>>>(Wv, Wvb, (int)NW, 1.0f);
  cvt_bf16<<<cG, cB, 0, stream>>>(Wo, Wob, (int)NW, 1.0f);

  dim3 gB(256), gG(DIMN/128, MROWS/128);
  gemm_nt_wmma<<<gG, gB, 0, stream>>>(Xq, Wqb, nullptr, Qp, 1);
  gemm_nt_wmma<<<gG, gB, 0, stream>>>(Xk, Wkb, nullptr, Kp, 1);
  gemm_nt_wmma<<<gG, gB, 0, stream>>>(Xv, Wvb, nullptr, Vt, 2);

  flash_attn<<<dim3(LSEQ/32, BATCH*NH), dim3(32), 0, stream>>>(Qp, Kp, Vt, Ctx);

  gemm_nt_wmma<<<gG, gB, 0, stream>>>(Ctx, Wob, out, nullptr, 0);
}
